// KNeighBallChanger_53017076302314
// MI455X (gfx1250) — matrix operand order repaired and checked by reading, not compiled, not run on hardware
//
#include <hip/hip_runtime.h>

// Ball-query mean pooling on gfx1250 (MI455X), fully tensor-core mapped.
//   ref: mask[o,i] = (|p_o - p_i|^2 <= r^2); out[b,o] = sum_i mask*x[b,i] / max(count,1)
//
// Distance test   -> 2x v_wmma_f32_16x16x4_f32 (G = I x O^T, -|i|^2/2 folded into K=3,
//                    per-column threshold (|o|^2-r^2)/2 is a loop-invariant lane scalar)
// Masked GEMM+cnt -> v_wmma_f32_16x16x32_f16 (A rows 0..7 = batches, row 8 = ones -> counts)
//
// Zero-shuffle layout bridge: the distance WMMAs' A operands are indexed so that
//   WMMA#0 rows 0-7 = inputs base+0..7,  rows 8-15 = inputs base+16..23
//   WMMA#1 rows 0-7 = inputs base+8..15, rows 8-15 = inputs base+24..31
// making each lane's own D registers exactly the 16 K-slots of its half of the
// f16 B matrix (lanes 0-15 hold K=0..15, lanes 16-31 hold K=16..31). No cross-lane ops.
//
// A2 constant rows (ones/zeros) are initialized once and preserved by exec-masked
// loads; dual accumulators break the WMMA serial chain across the 2x-unrolled body.
// K split 4-ways per out-tile for occupancy; partials combined via LDS in fixed order.

typedef __attribute__((ext_vector_type(2)))  float    v2f;
typedef __attribute__((ext_vector_type(8)))  float    v8f;
typedef __attribute__((ext_vector_type(16))) _Float16 v16h;

namespace {
constexpr int   kNIn      = 16384;
constexpr int   kNOut     = 8192;
constexpr int   kBatch    = 8;
constexpr float kRadius   = 0.05f;
constexpr int   kSlices   = 4;                 // waves cooperating per out-tile (K split)
constexpr int   kSliceLen = kNIn / kSlices;    // 4096 inputs per wave
constexpr int   kTilesPerBlock = 2;            // 8 waves/block = 2 tiles x 4 slices
}

// One 32-input chunk: 2 distance WMMAs -> mask pack -> masked-GEMM WMMA.
// A2 is a persistent register block: lanes n>=8 keep their constant rows because
// the loads inside `if (n < kBatch)` update it under exec mask only.
__device__ __forceinline__
v8f ball_chunk(const float* __restrict__ x, const float* __restrict__ ipos,
               int base, int n, bool hi, int offA, v2f bO, float thr,
               v16h& A2, v8f acc)
{
  // ---- A operands of the two distance WMMAs ---------------------------------
  // 32-bit A 16x4 layout: lanes 0-15 {K0,K1}, lanes 16-31 {K2,K3}, M = lane%16
  const int i0 = base + offA;
  const int i1 = i0 + 8;
  const float p0x = ipos[i0 * 3 + 0], p0y = ipos[i0 * 3 + 1], p0z = ipos[i0 * 3 + 2];
  const float p1x = ipos[i1 * 3 + 0], p1y = ipos[i1 * 3 + 1], p1z = ipos[i1 * 3 + 2];
  const float h0 = -0.5f * (p0x * p0x + p0y * p0y + p0z * p0z);
  const float h1 = -0.5f * (p1x * p1x + p1y * p1y + p1z * p1z);
  v2f a0; a0.x = hi ? p0z : p0x; a0.y = hi ? h0 : p0y;
  v2f a1; a1.x = hi ? p1z : p1x; a1.y = hi ? h1 : p1y;

  v8f zero = {};
  // D layout: VGPR r, lane l -> col o = l%16; row M = r + 8*(l>=16)
  v8f d0 = __builtin_amdgcn_wmma_f32_16x16x4_f32(false, a0, false, bO,
                                                 (short)0, zero, false, false);
  v8f d1 = __builtin_amdgcn_wmma_f32_16x16x4_f32(false, a1, false, bO,
                                                 (short)0, zero, false, false);

  // ---- threshold -> f16(1.0) bit patterns (exact in f16) --------------------
  unsigned m0[8], m1[8];
#pragma unroll
  for (int r = 0; r < 8; ++r) {
    m0[r] = (d0[r] >= thr) ? 0x3C00u : 0u;
    m1[r] = (d1[r] >= thr) ? 0x3C00u : 0u;
  }
  // ---- f16 B matrix (32x16): K packed 2-per-dword, identical in both halves --
  union { v16h h; unsigned u[8]; } Bm;
#pragma unroll
  for (int v = 0; v < 4; ++v)
    Bm.u[v] = m0[2 * v] | (m0[2 * v + 1] << 16);
#pragma unroll
  for (int v = 0; v < 4; ++v)
    Bm.u[4 + v] = m1[2 * v] | (m1[2 * v + 1] << 16);

  // ---- A = augmented x chunk (16x32 f16): masked overwrite of rows 0..7 ------
  if (n < kBatch) {
    const int g = hi ? 16 : 0;
    const float4* pa = reinterpret_cast<const float4*>(x + n * kNIn + base + g);
    const float4* pb = reinterpret_cast<const float4*>(x + n * kNIn + base + 8 + g);
    const float4 f0 = pa[0], f1 = pa[1], f2 = pb[0], f3 = pb[1];
    A2[0]  = (_Float16)f0.x; A2[1]  = (_Float16)f0.y;
    A2[2]  = (_Float16)f0.z; A2[3]  = (_Float16)f0.w;
    A2[4]  = (_Float16)f1.x; A2[5]  = (_Float16)f1.y;
    A2[6]  = (_Float16)f1.z; A2[7]  = (_Float16)f1.w;
    A2[8]  = (_Float16)f2.x; A2[9]  = (_Float16)f2.y;
    A2[10] = (_Float16)f2.z; A2[11] = (_Float16)f2.w;
    A2[12] = (_Float16)f3.x; A2[13] = (_Float16)f3.y;
    A2[14] = (_Float16)f3.z; A2[15] = (_Float16)f3.w;
  }

  return __builtin_amdgcn_wmma_f32_16x16x32_f16(false, A2, false, Bm.h,
                                                (short)0, acc, false, false);
}

__global__ __launch_bounds__(256)
void ball_pool_wmma(const float* __restrict__ x,      // [8, 16384]
                    const float* __restrict__ ipos,   // [16384, 3]
                    const float* __restrict__ opos,   // [8192, 3]
                    float* __restrict__ out)          // [8, 8192]
{
  __shared__ float lds[8][9][16];   // [wave][row(8 batches + count)][out-in-tile]

  const int  lane  = threadIdx.x & 31;
  const int  wid   = threadIdx.x >> 5;          // 0..7
  const int  slice = wid & (kSlices - 1);       // 0..3 : K slice
  const int  tIn   = wid / kSlices;             // 0..1 : tile within block
  const int  tile  = blockIdx.x * kTilesPerBlock + tIn;
  const int  obase = tile * 16;
  const int  n     = lane & 15;                 // out-point / matrix-row selector
  const bool hi    = lane >= 16;

  // A-row -> input-offset remap for the zero-shuffle layout bridge.
  const int offA = (n < 8) ? n : n + 8;

  // ---- loop-invariant per-lane out-point data --------------------------------
  const float ox = opos[(obase + n) * 3 + 0];
  const float oy = opos[(obase + n) * 3 + 1];
  const float oz = opos[(obase + n) * 3 + 2];
  // B operand of the distance WMMA (4x16 f32): K rows striped across VGPR+half.
  // lanes 0-15: {K0=xo, K1=yo}; lanes 16-31: {K2=zo, K3=1.0 (pairs with -|i|^2/2)}
  v2f bO;
  bO.x = hi ? oz   : ox;
  bO.y = hi ? 1.0f : oy;
  // mask[i,o] = (i.o - |i|^2/2 >= (|o|^2 - r^2)/2)  <=>  |i-o|^2 <= r^2
  const float thr = 0.5f * (ox * ox + oy * oy + oz * oz - kRadius * kRadius);

  // Persistent A2 blocks: row 8 = ones (counts), rows 9..15 = zeros; rows 0..7
  // are overwritten every chunk by the exec-masked loads.
  v16h A2a, A2b;
  {
    const _Float16 fill = (n == kBatch) ? (_Float16)1.0f : (_Float16)0.0f;
#pragma unroll
    for (int k = 0; k < 16; ++k) { A2a[k] = fill; A2b[k] = fill; }
  }

  v8f acc0 = {}, acc1 = {};                     // dual accumulators (even/odd chunk)
  const int kbeg = slice * kSliceLen;
  const int kend = kbeg + kSliceLen;

  for (int base = kbeg; base < kend; base += 64) {
    acc0 = ball_chunk(x, ipos, base,      n, hi, offA, bO, thr, A2a, acc0);
    acc1 = ball_chunk(x, ipos, base + 32, n, hi, offA, bO, thr, A2b, acc1);
  }
  const v8f acc = acc0 + acc1;

  // ---- combine the 4 K-slices deterministically via LDS ----------------------
  if (!hi) {
#pragma unroll
    for (int b = 0; b < 8; ++b) lds[wid][b][n] = acc[b];   // rows 0..7 (batch sums)
  } else {
    lds[wid][8][n] = acc[0];                               // row 8 (counts)
  }
  __syncthreads();

  if (slice == 0 && !hi) {
    float num[8] = {};
    float cnt = 0.0f;
#pragma unroll
    for (int s = 0; s < kSlices; ++s) {
#pragma unroll
      for (int b = 0; b < 8; ++b) num[b] += lds[wid + s][b][n];
      cnt += lds[wid + s][8][n];
    }
    const float dem = (cnt > 0.0f) ? cnt : 1.0f;
    const int o = obase + n;
#pragma unroll
    for (int b = 0; b < 8; ++b) out[b * kNOut + o] = num[b] / dem;
  }
}

extern "C" void kernel_launch(void* const* d_in, const int* in_sizes, int n_in,
                              void* d_out, int out_size, void* d_ws, size_t ws_size,
                              hipStream_t stream) {
  const float* x    = (const float*)d_in[0];   // [8, 16384]
  const float* ipos = (const float*)d_in[1];   // [16384, 3]
  const float* opos = (const float*)d_in[2];   // [8192, 3]
  float* out = (float*)d_out;                  // [8, 8192]

  const int tiles = kNOut / 16;                          // 512
  dim3 grid(tiles / kTilesPerBlock);                     // 256 blocks
  dim3 block(256);                                       // 8 waves: 2 tiles x 4 K-slices
  ball_pool_wmma<<<grid, block, 0, stream>>>(x, ipos, opos, out);
}
